// VisionTransformerEncoder_53042846105901
// MI455X (gfx1250) — matrix operand order repaired and checked by reading, not compiled, not run on hardware
//
#include <hip/hip_runtime.h>
#include <hip/hip_bf16.h>
#include <cstdint>
#include <math.h>

typedef __attribute__((ext_vector_type(16))) __bf16 v16bf;
typedef __attribute__((ext_vector_type(8)))  __bf16 v8bf;
typedef __attribute__((ext_vector_type(8)))  float  v8f;
typedef __attribute__((ext_vector_type(4)))  uint32_t v4u;
typedef __attribute__((ext_vector_type(8)))  int      v8i;
typedef __attribute__((ext_vector_type(4)))  int      v4i;

#define E_DIM   1024
#define N_HEADS 16
#define HD_DIM  64
#define MLP_DIM 4096
#define SEQ     1024
#define BATCH   8
#define ROWS    (BATCH * SEQ)   /* 8192 */

static __device__ __forceinline__ v16bf join16(v8bf a, v8bf b) {
    return __builtin_shufflevector(a, b, 0, 1, 2, 3, 4, 5, 6, 7,
                                         8, 9, 10, 11, 12, 13, 14, 15);
}
static __device__ __forceinline__ unsigned short bf16bits(float f) {
    __bf16 h = (__bf16)f;
    return __builtin_bit_cast(unsigned short, h);
}

// ---------------------------------------------------------------------------
// TDM: async-load a 64(M) x 32(K) bf16 tile (row stride K elements) from
// global into LDS, padding 4 dwords after every 16 dwords stored so rows land
// at a 40-half stride (bank-conflict-free fragment reads).
// D# layout per CDNA5 ISA.
// ---------------------------------------------------------------------------
static __device__ __forceinline__ void tdm_load_A(const __bf16* gA,
                                                  uint32_t ldsByte, int K) {
    uint64_t ga = (uint64_t)(uintptr_t)gA;
    v4u g0;
    g0[0] = 1u;                                   // count=1, user mode
    g0[1] = ldsByte;                              // lds_addr (bytes)
    g0[2] = (uint32_t)ga;                         // global_addr[31:0]
    g0[3] = (uint32_t)((ga >> 32) & 0x01ffffffu)  // global_addr[56:32]
            | (2u << 30);                         // type = 2 ("image")
    v8i g1;
    g1[0] = (int)((1u << 16)      // data_size = 1 -> 2 bytes
                | (1u << 20)      // pad_enable
                | (3u << 22)      // pad_interval: 16 dwords
                | (3u << 25));    // pad_amount:   4 dwords
    g1[1] = (int)(32u << 16);     // tensor_dim0 = 32  (bits 79:48 lo)
    g1[2] = (int)(64u << 16);     // tensor_dim1 = 64  (bits 111:80 lo)
    g1[3] = (int)(32u << 16);     // tile_dim0 = 32    (bits 127:112)
    g1[4] = (int)64u;             // tile_dim1 = 64, tile_dim2 = 0
    g1[5] = (int)(uint32_t)K;     // tensor_dim0_stride lo32 (elements)
    g1[6] = 0;
    g1[7] = 0;
    v4i z = {0, 0, 0, 0};
#if __clang_major__ >= 23
    v8i z8 = {0, 0, 0, 0, 0, 0, 0, 0};
    __builtin_amdgcn_tensor_load_to_lds(g0, g1, z, z, z8, 0);
#else
    __builtin_amdgcn_tensor_load_to_lds(g0, g1, z, z, 0);
#endif
}

// ---------------------------------------------------------------------------
// fp32 -> bf16 elementwise conversion (weights staging)
// ---------------------------------------------------------------------------
__global__ void cvt_f32_bf16(const float* __restrict__ in,
                             __bf16* __restrict__ out, long n) {
    long i = (long)blockIdx.x * blockDim.x + threadIdx.x;
    long stride = (long)gridDim.x * blockDim.x;
    for (; i < n; i += stride) out[i] = (__bf16)in[i];
}

// ---------------------------------------------------------------------------
// LayerNorm (fp32 in, bf16 out), one block (256 thr) per row of E_DIM
// ---------------------------------------------------------------------------
__global__ __launch_bounds__(256)
void ln_bf16(const float* __restrict__ x, const float* __restrict__ g,
             const float* __restrict__ b, __bf16* __restrict__ out) {
    __shared__ float red[256];
    int row = blockIdx.x;
    const float* xr = x + (size_t)row * E_DIM;
    int t = threadIdx.x;

    float s = 0.f;
    for (int i = t; i < E_DIM; i += 256) s += xr[i];
    red[t] = s; __syncthreads();
    for (int o = 128; o > 0; o >>= 1) { if (t < o) red[t] += red[t + o]; __syncthreads(); }
    float mu = red[0] * (1.0f / E_DIM);
    __syncthreads();

    float v = 0.f;
    for (int i = t; i < E_DIM; i += 256) { float d = xr[i] - mu; v += d * d; }
    red[t] = v; __syncthreads();
    for (int o = 128; o > 0; o >>= 1) { if (t < o) red[t] += red[t + o]; __syncthreads(); }
    float rstd = rsqrtf(red[0] * (1.0f / E_DIM) + 1e-6f);

    __bf16* orow = out + (size_t)row * E_DIM;
    for (int i = t; i < E_DIM; i += 256)
        orow[i] = (__bf16)((xr[i] - mu) * rstd * g[i] + b[i]);
}

// ---------------------------------------------------------------------------
// Tiled bf16 WMMA GEMM: C[M,N] = A[M,K] @ B[K,N] + bias (+resid) (opt GELU)
// 128 threads = 4 waves; block tile 64x64; K stepped by 32.
// Double-buffered: A tiles fetched by the Tensor Data Mover (async, TENSORcnt),
// B tiles staged transposed by VALU stores, next tiles prefetched during
// current compute. Fragments are two ds_load_b128 each.
// ---------------------------------------------------------------------------
__global__ __launch_bounds__(128)
void gemm_bf16(const __bf16* __restrict__ A, const __bf16* __restrict__ B,
               const float* __restrict__ bias, const float* __restrict__ resid,
               float* __restrict__ outF, __bf16* __restrict__ outB,
               int M, int N, int K, int gelu) {
    __shared__ __attribute__((aligned(16))) unsigned short As2[2][64 * 40];
    __shared__ __attribute__((aligned(16))) unsigned short Bt2[2][64 * 40];

    int t = threadIdx.x;
    int lane = t & 31, wid = t >> 5;
    int l15 = lane & 15, hi = lane >> 4;
    int n0 = blockIdx.x * 64, m0 = blockIdx.y * 64;

    int rA   = wid * 16 + l15;
    int koff = hi * 8;     // A-fragment K sub-offset (ISA 16-bit A layout)
    int kk0  = hi * 16;    // unused; B fragments also use koff

    // stage B tile 32x64 at k0, transposed into Bt2[buf][n][k]
    auto stage_b = [&](int buf, int k0) {
        unsigned short* Bt = &Bt2[buf][0];
#pragma unroll
        for (int i = 0; i < 8; ++i) {
            int p = i * 128 + t;
            int r = p >> 5;             // k row 0..31
            int c = (p & 31) * 2;       // n col pair
            uint32_t w = *(const uint32_t*)(B + (size_t)(k0 + r) * N + n0 + c);
            Bt[c * 40 + r]       = (unsigned short)(w & 0xffffu);
            Bt[(c + 1) * 40 + r] = (unsigned short)(w >> 16);
        }
    };

    v8f acc[4] = {};
    int nsteps = K / 32;

    // prologue: fetch step-0 tiles
    if (wid == 0)
        tdm_load_A(A + (size_t)m0 * K,
                   (uint32_t)(uintptr_t)(void*)&As2[0][0], K);
    stage_b(0, 0);
    __builtin_amdgcn_s_wait_tensorcnt(0);
    __syncthreads();

    for (int s = 0; s < nsteps; ++s) {
        int cur = s & 1, nxt = cur ^ 1;

        // prefetch next tiles (overlaps with compute below)
        if (s + 1 < nsteps) {
            if (wid == 0)
                tdm_load_A(A + (size_t)m0 * K + (s + 1) * 32,
                           (uint32_t)(uintptr_t)(void*)&As2[nxt][0], K);
            stage_b(nxt, (s + 1) * 32);
        }

        // ---- A fragment: two b128 LDS loads ----
        const unsigned short* As = &As2[cur][0];
        const unsigned short* Bt = &Bt2[cur][0];
        v8bf a_lo = *(const v8bf*)(&As[rA * 40 + koff]);
        v8bf a_hi = *(const v8bf*)(&As[rA * 40 + koff + 16]);
        v16bf af = join16(a_lo, a_hi);

        // ---- B fragments + WMMA ----
#pragma unroll
        for (int nt = 0; nt < 4; ++nt) {
            int c = nt * 16 + l15;
            v8bf b_lo = *(const v8bf*)(&Bt[c * 40 + koff]);
            v8bf b_hi = *(const v8bf*)(&Bt[c * 40 + koff + 16]);
            v16bf bf = join16(b_lo, b_hi);
            acc[nt] = __builtin_amdgcn_wmma_f32_16x16x32_bf16(
                false, af, false, bf, (short)0, acc[nt], false, false);
        }

        __builtin_amdgcn_s_wait_tensorcnt(0);
        __syncthreads();
    }

    // ---- epilogue: C/D layout -> global ----
    int mbase = m0 + wid * 16 + hi * 8;
#pragma unroll
    for (int nt = 0; nt < 4; ++nt) {
        int n = n0 + nt * 16 + l15;
        float bn = bias[n];
#pragma unroll
        for (int r = 0; r < 8; ++r) {
            int m = mbase + r;
            float v = acc[nt][r] + bn;
            if (gelu) v = 0.5f * v * (1.0f + erff(v * 0.70710678118654752f));
            if (resid) v += resid[(size_t)m * N + n];
            if (outF) outF[(size_t)m * N + n] = v;
            if (outB) outB[(size_t)m * N + n] = (__bf16)v;
        }
    }
}

// ---------------------------------------------------------------------------
// Attention: one wave per (batch, head, 16-query tile).
// qkv row = b*SEQ+n (3072 wide): q @ [h*64+d], k @ [1024+..], v @ [2048+..]
// Q/K fragments: contiguous global b128 loads. V staged transposed in LDS.
// Probs packed bf16 in padded LDS for b128 fragment loads.
// ---------------------------------------------------------------------------
#define PBS 1032   /* prob row stride in halves: 16B-aligned, 4-bank lane step */
__global__ __launch_bounds__(32)
void attn_kernel(const __bf16* __restrict__ qkv, __bf16* __restrict__ ctx) {
    __shared__ float sc[16 * SEQ];                                       // 64 KB
    __shared__ __attribute__((aligned(16))) unsigned short pb[16 * PBS]; // 33 KB
    __shared__ __attribute__((aligned(16))) unsigned short Vt[64 * 40];  // 5 KB

    int lane = threadIdx.x;
    int l15 = lane & 15, hi = lane >> 4;
    int qt = blockIdx.x, h = blockIdx.y, b = blockIdx.z;
    int koff = hi * 8;
    int kk0  = hi * 16;

    // ---- Q fragments: 16x64 as two 16x32 A fragments (global b128 loads) ----
    size_t qrow = ((size_t)b * SEQ + qt * 16 + l15) * 3072 + h * HD_DIM;
    v16bf qf[2];
#pragma unroll
    for (int kc = 0; kc < 2; ++kc) {
        v8bf lo = *(const v8bf*)(qkv + qrow + kc * 32 + koff);
        v8bf hh = *(const v8bf*)(qkv + qrow + kc * 32 + koff + 16);
        qf[kc] = join16(lo, hh);
    }

    // ---- scores = Q @ K^T / 8 ----
    const float scale = 0.125f;   // 1/sqrt(64)
    for (int nt = 0; nt < 64; ++nt) {
        int key = nt * 16 + l15;
        size_t krow = ((size_t)b * SEQ + key) * 3072 + E_DIM + h * HD_DIM;
        v8f acc = {};
#pragma unroll
        for (int kc = 0; kc < 2; ++kc) {
            v8bf lo = *(const v8bf*)(qkv + krow + kc * 32 + koff);
            v8bf hh = *(const v8bf*)(qkv + krow + kc * 32 + koff + 16);
            v16bf kf = join16(lo, hh);
            acc = __builtin_amdgcn_wmma_f32_16x16x32_bf16(
                false, qf[kc], false, kf, (short)0, acc, false, false);
        }
#pragma unroll
        for (int r = 0; r < 8; ++r)
            sc[(r + hi * 8) * SEQ + nt * 16 + l15] = acc[r] * scale;
    }
    __syncthreads();

    // ---- softmax: 2 lanes per row (halves joined via shfl_xor) ----
    {
        float* rp = sc + l15 * SEQ + hi * 512;
        float mx = -1e30f;
        for (int i = 0; i < 512; ++i) mx = fmaxf(mx, rp[i]);
        mx = fmaxf(mx, __shfl_xor(mx, 16, 32));
        float sum = 0.f;
        for (int i = 0; i < 512; ++i) { float e = expf(rp[i] - mx); rp[i] = e; sum += e; }
        sum += __shfl_xor(sum, 16, 32);
        float inv = 1.0f / sum;
        unsigned short* pp = pb + l15 * PBS + hi * 512;
        for (int i = 0; i < 512; ++i) pp[i] = bf16bits(rp[i] * inv);
    }
    __syncthreads();

    // ---- context = P @ V  (K=1024 in 32-key chunks; V staged transposed) ----
    v8f acc[4] = {};
    for (int kc = 0; kc < 32; ++kc) {
        // stage V chunk (32 keys x 64 hd) transposed: Vt[hd][key_local]
#pragma unroll
        for (int i = 0; i < 32; ++i) {
            int p = i * 32 + lane;
            int r = p >> 5;            // key_local 0..31
            int c = (p & 31) * 2;      // hd pair
            uint32_t w = *(const uint32_t*)(qkv +
                ((size_t)b * SEQ + kc * 32 + r) * 3072 + 2 * E_DIM + h * HD_DIM + c);
            Vt[c * 40 + r]       = (unsigned short)(w & 0xffffu);
            Vt[(c + 1) * 40 + r] = (unsigned short)(w >> 16);
        }
        __syncthreads();

        // P fragment: two b128 LDS loads from packed bf16 probs
        v8bf p_lo = *(const v8bf*)(&pb[l15 * PBS + kc * 32 + koff]);
        v8bf p_hi = *(const v8bf*)(&pb[l15 * PBS + kc * 32 + koff + 16]);
        v16bf pf = join16(p_lo, p_hi);

#pragma unroll
        for (int nt = 0; nt < 4; ++nt) {
            int c = nt * 16 + l15;
            v8bf v_lo = *(const v8bf*)(&Vt[c * 40 + koff]);
            v8bf v_hi = *(const v8bf*)(&Vt[c * 40 + koff + 16]);
            v16bf vf = join16(v_lo, v_hi);
            acc[nt] = __builtin_amdgcn_wmma_f32_16x16x32_bf16(
                false, pf, false, vf, (short)0, acc[nt], false, false);
        }
        __syncthreads();
    }

#pragma unroll
    for (int nt = 0; nt < 4; ++nt)
#pragma unroll
        for (int r = 0; r < 8; ++r) {
            int m = qt * 16 + r + hi * 8;
            ctx[((size_t)b * SEQ + m) * E_DIM + h * HD_DIM + nt * 16 + l15] =
                (__bf16)acc[nt][r];
        }
}

// ---------------------------------------------------------------------------
// Orchestration
// ---------------------------------------------------------------------------
extern "C" void kernel_launch(void* const* d_in, const int* in_sizes, int n_in,
                              void* d_out, int out_size, void* d_ws, size_t ws_size,
                              hipStream_t stream) {
    (void)in_sizes; (void)n_in; (void)out_size; (void)ws_size;

    const float* x     = (const float*)d_in[0];
    const float* ln1_g = (const float*)d_in[1];
    const float* ln1_b = (const float*)d_in[2];
    const float* w_qkv = (const float*)d_in[3];
    const float* b_qkv = (const float*)d_in[4];
    const float* w_o   = (const float*)d_in[5];
    const float* b_o   = (const float*)d_in[6];
    const float* ln2_g = (const float*)d_in[7];
    const float* ln2_b = (const float*)d_in[8];
    const float* w_fc1 = (const float*)d_in[9];
    const float* b_fc1 = (const float*)d_in[10];
    const float* w_fc2 = (const float*)d_in[11];
    const float* b_fc2 = (const float*)d_in[12];
    float* outF = (float*)d_out;

    // workspace layout (bf16 elements)
    __bf16* ws = (__bf16*)d_ws;
    size_t o = 0;
    __bf16* wsQKV  = ws + o; o += (size_t)E_DIM * 3 * E_DIM;
    __bf16* wsO    = ws + o; o += (size_t)E_DIM * E_DIM;
    __bf16* wsF1   = ws + o; o += (size_t)E_DIM * MLP_DIM;
    __bf16* wsF2   = ws + o; o += (size_t)MLP_DIM * E_DIM;
    __bf16* wsXN   = ws + o; o += (size_t)ROWS * E_DIM;
    __bf16* wsQKVo = ws + o; o += (size_t)ROWS * 3 * E_DIM;
    __bf16* wsCTX  = ws + o; o += (size_t)ROWS * E_DIM;
    __bf16* wsH    = ws + o; o += (size_t)ROWS * MLP_DIM;

    // 1) stage weights to bf16
    cvt_f32_bf16<<<2048, 256, 0, stream>>>(w_qkv, wsQKV, (long)E_DIM * 3 * E_DIM);
    cvt_f32_bf16<<<2048, 256, 0, stream>>>(w_o,   wsO,   (long)E_DIM * E_DIM);
    cvt_f32_bf16<<<2048, 256, 0, stream>>>(w_fc1, wsF1,  (long)E_DIM * MLP_DIM);
    cvt_f32_bf16<<<2048, 256, 0, stream>>>(w_fc2, wsF2,  (long)MLP_DIM * E_DIM);

    // 2) LN1 -> bf16
    ln_bf16<<<ROWS, 256, 0, stream>>>(x, ln1_g, ln1_b, wsXN);

    // 3) QKV GEMM: (8192x1024)@(1024x3072) + b_qkv -> bf16
    gemm_bf16<<<dim3(3 * E_DIM / 64, ROWS / 64), 128, 0, stream>>>(
        wsXN, wsQKV, b_qkv, nullptr, nullptr, wsQKVo,
        ROWS, 3 * E_DIM, E_DIM, 0);

    // 4) attention
    attn_kernel<<<dim3(SEQ / 16, N_HEADS, BATCH), 32, 0, stream>>>(wsQKVo, wsCTX);

    // 5) O-proj + residual -> x1 (fp32 in d_out)
    gemm_bf16<<<dim3(E_DIM / 64, ROWS / 64), 128, 0, stream>>>(
        wsCTX, wsO, b_o, x, outF, nullptr,
        ROWS, E_DIM, E_DIM, 0);

    // 6) LN2 -> bf16
    ln_bf16<<<ROWS, 256, 0, stream>>>(outF, ln2_g, ln2_b, wsXN);

    // 7) FC1 + GELU -> bf16
    gemm_bf16<<<dim3(MLP_DIM / 64, ROWS / 64), 128, 0, stream>>>(
        wsXN, wsF1, b_fc1, nullptr, nullptr, wsH,
        ROWS, MLP_DIM, E_DIM, 1);

    // 8) FC2 + residual (read+write d_out, thread-local RMW) -> fp32 out
    gemm_bf16<<<dim3(E_DIM / 64, ROWS / 64), 128, 0, stream>>>(
        wsH, wsF2, b_fc2, outF, outF, nullptr,
        ROWS, E_DIM, MLP_DIM, 0);
}
